// ASTRAMoE_44770739094071
// MI455X (gfx1250) — matrix operand order repaired and checked by reading, not compiled, not run on hardware
//
#include <hip/hip_runtime.h>
#include <stdint.h>

#define N_TOK   262144
#define D_DIM   256
#define E_NUM   4
#define C_NUM   10
#define MTILE   128
#define NBLK    (N_TOK / MTILE)   // 2048
#define NSEG    6                 // gate, expert0..3, alpha

typedef __attribute__((ext_vector_type(16))) __bf16 v16bf;
typedef __attribute__((ext_vector_type(8)))  float  v8f;
typedef __attribute__((ext_vector_type(8)))  unsigned short u16x8;

union Frag {
  v16bf bf;
  struct { u16x8 lo, hi; } p;
};

__device__ __forceinline__ uint16_t f2bf(float f) {
  union { float f; uint32_t u; } cv; cv.f = f;
  uint32_t u = cv.u;
  uint32_t r = u + 0x7FFFu + ((u >> 16) & 1u);   // round-to-nearest-even
  return (uint16_t)(r >> 16);
}

// branch-free erf (Abramowitz & Stegun 7.1.26, |err| <= 1.5e-7)
__device__ __forceinline__ float erf_fast(float x) {
  const float ax = fabsf(x);
  const float t  = __builtin_amdgcn_rcpf(fmaf(0.3275911f, ax, 1.0f));
  float p = fmaf(1.061405429f, t, -1.453152027f);
  p = fmaf(p, t, 1.421413741f);
  p = fmaf(p, t, -0.284496736f);
  p = fmaf(p, t, 0.254829592f);
  p *= t;
  const float e  = __expf(-ax * ax);
  const float er = fmaf(-p, e, 1.0f);
  return copysignf(er, x);
}

__device__ __forceinline__ float gelu_exact(float x) {
  return 0.5f * x * (1.0f + erf_fast(x * 0.70710678118f));
}

// branch-free softplus: max(x,0) + log1p(exp(-|x|))
__device__ __forceinline__ float softplus_fast(float x) {
  return fmaxf(x, 0.0f) + __logf(1.0f + __expf(-fabsf(x)));
}

// ---------------------------------------------------------------------------
// Prep: pack layer-1 weights -> W1T[1536][256] bf16 (transposed, WMMA-B ready)
//       pack layer-2 heads   -> W2T[96][256]  bf16 (cols padded 4/10 -> 16)
//       fused biases b1[1536], b2[96]
// grid = 1536 + 96 = 1632 blocks, 256 threads (thread = k index)
// ---------------------------------------------------------------------------
__global__ void prep_pack(const float* __restrict__ Wg1, const float* __restrict__ Wg2,
                          const float* __restrict__ We1, const float* __restrict__ We2,
                          const float* __restrict__ Wa1, const float* __restrict__ Wa2,
                          const float* __restrict__ bg1, const float* __restrict__ bg2,
                          const float* __restrict__ be1, const float* __restrict__ be2,
                          const float* __restrict__ ba1, const float* __restrict__ ba2,
                          uint16_t* __restrict__ W1T, uint16_t* __restrict__ W2T,
                          float* __restrict__ b1, float* __restrict__ b2)
{
  const int bid = blockIdx.x;
  const int k   = threadIdx.x;          // 0..255
  if (bid < 1536) {
    const int j = bid;                  // global layer-1 output column
    float v, bv;
    if (j < 256)        { v = Wg1[k * 256 + j];                      bv = bg1[j]; }
    else if (j < 1280)  { int e = (j - 256) >> 8, jj = (j - 256) & 255;
                          v = We1[e * 65536 + k * 256 + jj];          bv = be1[e * 256 + jj]; }
    else                { int jj = j - 1280;
                          v = Wa1[k * 256 + jj];                      bv = ba1[jj]; }
    W1T[((size_t)j << 8) + k] = f2bf(v);
    if (k == 0) b1[j] = bv;
  } else {
    const int idx = bid - 1536;         // 0..95
    const int s = idx >> 4, c = idx & 15;
    float v = 0.0f, bv = 0.0f;
    if (s == 0)      { if (c < E_NUM) { v = Wg2[k * E_NUM + c];                bv = bg2[c]; } }
    else if (s <= 4) { int e = s - 1;
                       if (c < C_NUM) { v = We2[e * 2560 + k * C_NUM + c];     bv = be2[e * C_NUM + c]; } }
    else             { if (c < C_NUM) { v = Wa2[k * C_NUM + c];                bv = ba2[c]; } }
    W2T[((size_t)idx << 8) + k] = f2bf(v);
    if (k == 0) b2[idx] = bv;
  }
}

// cooperative 16-col x 256-K bf16 B-tile load into LDS (32 B per thread)
__device__ __forceinline__ void load_b_tile(uint16_t (*sB)[264],
                                            const uint16_t* __restrict__ W,
                                            int rowBase)
{
  const int t  = threadIdx.x;
  const int c  = t >> 4;            // 0..15 column within tile
  const int kk = (t & 15) << 4;     // 0..240, 16 halves
  const uint4* src = (const uint4*)(W + (((size_t)(rowBase + c)) << 8) + kk);
  uint4 a = src[0], b = src[1];
  *(uint4*)&sB[c][kk]     = a;
  *(uint4*)&sB[c][kk + 8] = b;
}

__device__ __forceinline__ void load_frag(Frag& a, Frag& b,
                                          const uint16_t (*sA)[264],
                                          const uint16_t (*sBt)[264],
                                          int rowA, int lcol, int half, int kt)
{
  const int ka = kt * 32 + half * 8;     // A: K 0-7/16-23 vs 8-15/24-31
  a.p.lo = *(const u16x8*)&sA[rowA][ka];
  a.p.hi = *(const u16x8*)&sA[rowA][ka + 16];
  const int kb = kt * 32 + half * 8;     // B: K 0-7/16-23 vs 8-15/24-31
  b.p.lo = *(const u16x8*)&sBt[lcol][kb];
  b.p.hi = *(const u16x8*)&sBt[lcol][kb + 16];
}

// 16x16x256 GEMM: software-pipelined (frags for kt+1 in flight during WMMA kt)
__device__ __forceinline__ v8f gemm_k256(const uint16_t (*sA)[264],
                                         const uint16_t (*sBt)[264],
                                         int rowA, int lcol, int half, v8f acc)
{
  Frag aC, bC, aN, bN;
  load_frag(aC, bC, sA, sBt, rowA, lcol, half, 0);
#pragma unroll
  for (int kt = 0; kt < 8; ++kt) {
    if (kt < 7) load_frag(aN, bN, sA, sBt, rowA, lcol, half, kt + 1);
    acc = __builtin_amdgcn_wmma_f32_16x16x32_bf16(false, aC.bf, false, bC.bf,
                                                  (short)0, acc, false, false);
    aC = aN; bC = bN;   // SSA rename under full unroll, no v_mov cost
  }
  return acc;
}

// ---------------------------------------------------------------------------
// Fused MoE forward. 2048 blocks x 256 threads (8 waves), 128 tokens/block.
// ---------------------------------------------------------------------------
__global__ __launch_bounds__(256) void moe_main(
    const float* __restrict__ x,
    const uint16_t* __restrict__ W1T,
    const uint16_t* __restrict__ W2T,
    const float* __restrict__ b1,
    const float* __restrict__ b2,
    float* __restrict__ outLogits,
    float* __restrict__ outAlpha,
    float* __restrict__ outGW,
    float* __restrict__ loadPart)
{
  __shared__ uint16_t sX[MTILE][264];     // x tile, bf16, padded stride
  __shared__ uint16_t sH[MTILE][264];     // hidden activations, bf16
  __shared__ float    sGW[MTILE][E_NUM];  // gate weights
  __shared__ float    sB1[1536];          // fused layer-1 bias
  __shared__ float    sB2[96];            // fused layer-2 bias
  __shared__ __align__(16) char sScratch[16 * 264 * 2];  // sB / sTmp overlay
  uint16_t (*sB)[264] = (uint16_t (*)[264])sScratch;     // 8448 B
  float    (*sTmp)[16] = (float (*)[16])sScratch;        // 8192 B (disjoint lifetime)

  const int blk  = blockIdx.x;
  const int row0 = blk * MTILE;
  const int tid  = threadIdx.x;
  const int w    = tid >> 5;        // wave 0..7 -> row block
  const int lane = tid & 31;        // wave32
  const int half = lane >> 4;
  const int lcol = lane & 15;       // N column of WMMA tile
  const int rowA = w * 16 + lcol;   // A-fragment source row

  // ---- stage X tile to LDS (fp32 -> bf16) + bias preload ----
  {
    const int r  = tid >> 1;
    const int cb = (tid & 1) << 7;
    const float4* src = (const float4*)(x + (size_t)(row0 + r) * D_DIM + cb);
#pragma unroll
    for (int j = 0; j < 32; ++j) {
      float4 v = src[j];
      const int c = cb + j * 4;
      sX[r][c + 0] = f2bf(v.x); sX[r][c + 1] = f2bf(v.y);
      sX[r][c + 2] = f2bf(v.z); sX[r][c + 3] = f2bf(v.w);
    }
#pragma unroll
    for (int i = 0; i < 6; ++i) sB1[tid + 256 * i] = b1[tid + 256 * i];
    if (tid < 96) sB2[tid] = b2[tid];
  }

  v8f logitsAcc = {};

  for (int s = 0; s < NSEG; ++s) {
    const int colBase = s * 256;

    // ---- layer 1: hidden = GELU(X @ W1seg + b1seg), 16 column tiles ----
    for (int ct = 0; ct < 16; ++ct) {
      __syncthreads();
      load_b_tile(sB, W1T, colBase + ct * 16);
      if (ct < 15)  // prefetch next weight tile into cache hierarchy
        __builtin_prefetch(W1T + (((size_t)(colBase + (ct + 1) * 16 + (tid >> 4))) << 8), 0, 1);
      __syncthreads();

      const float bias = sB1[colBase + ct * 16 + lcol];
      v8f acc = {bias, bias, bias, bias, bias, bias, bias, bias};
      acc = gemm_k256(sX, (const uint16_t (*)[264])sB, rowA, lcol, half, acc);

#pragma unroll
      for (int v = 0; v < 8; ++v)           // C: M = v + 8*half, N = lcol
        sH[w * 16 + v + 8 * half][ct * 16 + lcol] = f2bf(gelu_exact(acc[v]));
    }

    // ---- layer 2: out[128x16] = hidden @ W2seg + b2seg ----
    __syncthreads();
    load_b_tile(sB, W2T, s * 16);
    __syncthreads();

    const float bias2 = sB2[s * 16 + lcol];
    v8f acc2 = {bias2, bias2, bias2, bias2, bias2, bias2, bias2, bias2};
    acc2 = gemm_k256((const uint16_t (*)[264])sH, (const uint16_t (*)[264])sB,
                     rowA, lcol, half, acc2);

    if (s == 0) {
      // ---- gate: top-2 + softmax over [v1, v2, 0, 0] layout ----
      __syncthreads();   // all waves done reading sB before sTmp overlay write
#pragma unroll
      for (int v = 0; v < 8; ++v) sTmp[w * 16 + v + 8 * half][lcol] = acc2[v];
      __syncthreads();
      if (tid < MTILE) {
        float gv[4] = { sTmp[tid][0], sTmp[tid][1], sTmp[tid][2], sTmp[tid][3] };
        int i1 = 0; float m1 = gv[0];
#pragma unroll
        for (int e = 1; e < 4; ++e) if (gv[e] > m1) { m1 = gv[e]; i1 = e; }
        int i2 = -1; float m2 = -3.4e38f;
#pragma unroll
        for (int e = 0; e < 4; ++e) if (e != i1 && gv[e] > m2) { m2 = gv[e]; i2 = e; }
        const float mx = fmaxf(m1, 0.0f);     // zeros participate in softmax
        float z[4], sum = 0.0f;
#pragma unroll
        for (int e = 0; e < 4; ++e) {
          z[e] = ((e == i1) || (e == i2)) ? __expf(gv[e] - mx) : __expf(-mx);
          sum += z[e];
        }
        const float inv = 1.0f / sum;
#pragma unroll
        for (int e = 0; e < 4; ++e) {
          float wgt = z[e] * inv;
          sGW[tid][e] = wgt;
          outGW[(size_t)(row0 + tid) * E_NUM + e] = wgt;
        }
      }
      __syncthreads();
    } else if (s <= 4) {
      // ---- expert e: logits += gw[:,e] * expert_out ----
      const int e = s - 1;
#pragma unroll
      for (int v = 0; v < 8; ++v)
        logitsAcc[v] += sGW[w * 16 + v + 8 * half][e] * acc2[v];
    } else {
      // ---- alpha: softplus + 1e-6 ----
      if (lcol < C_NUM) {
#pragma unroll
        for (int v = 0; v < 8; ++v)
          outAlpha[(size_t)(row0 + w * 16 + v + 8 * half) * C_NUM + lcol] =
              softplus_fast(acc2[v]) + 1e-6f;
      }
    }
  }

  // ---- write logits ----
  if (lcol < C_NUM) {
#pragma unroll
    for (int v = 0; v < 8; ++v)
      outLogits[(size_t)(row0 + w * 16 + v + 8 * half) * C_NUM + lcol] = logitsAcc[v];
  }

  // ---- deterministic block-local load reduction ----
  __syncthreads();
  if (tid < E_NUM) {
    float sum = 0.0f;
    for (int t = 0; t < MTILE; ++t) sum += sGW[t][tid];
    loadPart[(size_t)blk * E_NUM + tid] = sum;
  }
}

// fixed-order final reduction of per-block load partials
__global__ void reduce_load(const float* __restrict__ part, float* __restrict__ outLoad) {
  const int e = threadIdx.x;
  if (e < E_NUM) {
    float s = 0.0f;
    for (int b = 0; b < NBLK; ++b) s += part[(size_t)b * E_NUM + e];
    outLoad[e] = s;
  }
}

extern "C" void kernel_launch(void* const* d_in, const int* in_sizes, int n_in,
                              void* d_out, int out_size, void* d_ws, size_t ws_size,
                              hipStream_t stream) {
  (void)in_sizes; (void)n_in; (void)out_size; (void)ws_size;
  const float* x   = (const float*)d_in[0];
  const float* Wg1 = (const float*)d_in[1];
  const float* bg1 = (const float*)d_in[2];
  const float* Wg2 = (const float*)d_in[3];
  const float* bg2 = (const float*)d_in[4];
  const float* We1 = (const float*)d_in[5];
  const float* be1 = (const float*)d_in[6];
  const float* We2 = (const float*)d_in[7];
  const float* be2 = (const float*)d_in[8];
  const float* Wa1 = (const float*)d_in[9];
  const float* ba1 = (const float*)d_in[10];
  const float* Wa2 = (const float*)d_in[11];
  const float* ba2 = (const float*)d_in[12];

  char* ws = (char*)d_ws;
  uint16_t* W1T = (uint16_t*)(ws);             // 1536*256*2 = 786432 B
  uint16_t* W2T = (uint16_t*)(ws + 786432);    //   96*256*2 =  49152 B
  float*    b1  = (float*)   (ws + 835584);    // 1536*4     =   6144 B
  float*    b2  = (float*)   (ws + 841728);    //   96*4     =    384 B
  float*    lp  = (float*)   (ws + 842112);    // 2048*4*4   =  32768 B

  float* out       = (float*)d_out;
  float* outLogits = out;                                     // [N, 10]
  float* outAlpha  = out + (size_t)N_TOK * C_NUM;             // [N, 10]
  float* outGW     = out + (size_t)2 * N_TOK * C_NUM;         // [N, 4]
  float* outLoad   = outGW + (size_t)N_TOK * E_NUM;           // [4]

  prep_pack<<<1632, 256, 0, stream>>>(Wg1, Wg2, We1, We2, Wa1, Wa2,
                                      bg1, bg2, be1, be2, ba1, ba2,
                                      W1T, W2T, b1, b2);
  moe_main<<<NBLK, 256, 0, stream>>>(x, W1T, W2T, b1, b2,
                                     outLogits, outAlpha, outGW, lp);
  reduce_load<<<1, 32, 0, stream>>>(lp, outLoad);
}
